// DeepSeekV3_4879082848968
// MI455X (gfx1250) — matrix operand order repaired and checked by reading, not compiled, not run on hardware
//
#include <hip/hip_runtime.h>
#include <hip/hip_bf16.h>
#include <math.h>

// ---------------- types ----------------
typedef __bf16 bf16;
typedef __attribute__((ext_vector_type(16))) __bf16 bf16x16;
typedef __attribute__((ext_vector_type(8)))  __bf16 bf16x8;
typedef __attribute__((ext_vector_type(8)))  float  f32x8;
typedef __attribute__((ext_vector_type(4)))  int    v4i;

__device__ __forceinline__ f32x8 zero8() {
    f32x8 z; for (int i = 0; i < 8; ++i) z[i] = 0.f; return z;
}

#define WMMA_BF16(a, b, c) \
    __builtin_amdgcn_wmma_f32_16x16x32_bf16(false, (a), false, (b), (short)0, (c), false, false)

// ---------------- CDNA5 async global->LDS copy (ASYNCcnt path) ----------------
#if defined(__has_builtin)
#  if __has_builtin(__builtin_amdgcn_global_load_async_to_lds_b128)
#    define HAVE_ASYNC_LDS 1
#  endif
#endif
#ifndef HAVE_ASYNC_LDS
#  define HAVE_ASYNC_LDS 0
#endif

__device__ __forceinline__ void async_copy_b128(void* lds_dst, const void* gsrc) {
#if HAVE_ASYNC_LDS
    __builtin_amdgcn_global_load_async_to_lds_b128(
        (__attribute__((address_space(1))) v4i*)gsrc,
        (__attribute__((address_space(3))) v4i*)lds_dst, 0, 0);
#else
    *(bf16x8*)lds_dst = *(const bf16x8*)gsrc;
#endif
}

__device__ __forceinline__ void async_wait() {
#if HAVE_ASYNC_LDS
#  if __has_builtin(__builtin_amdgcn_s_wait_asynccnt)
    __builtin_amdgcn_s_wait_asynccnt(0);
#  else
    asm volatile("s_wait_asynccnt 0x0" ::: "memory");
#  endif
#endif
}

// ---------------- problem constants ----------------
constexpr int Bb = 4, Ss = 1024, Dd = 1024, Hh = 4, Ee = 32, Ff = 512, Kk = 8, NCc = 10;
constexpr int Tt = Bb * Ss;   // 4096 tokens
constexpr int DK = 256;       // per-head dim == compressed KV dim
constexpr float EPSc = 1e-6f;

// =====================================================================
// Tiled WMMA GEMM: C[M,N] = rowScale*(A[M,K]bf16 @ B[K,N]f32->bf16) + bias (+C)
// Block: 256 threads = 8 waves. Tile: 128(M) x 128(N), K-step 32.
// Double-buffered LDS; A staged via async global->LDS b128; B staged with
// coalesced f32 reads + on-the-fly bf16 convert, written transposed [n][k].
// Wave w computes rows [w*16, w*16+16) of the tile, all 128 cols (8 accs).
// =====================================================================
__global__ __launch_bounds__(256)
void k_gemm(const bf16* __restrict__ A, const float* __restrict__ Bm,
            float* __restrict__ Cf, bf16* __restrict__ Cb,
            int M, int N, int Kd,
            const float* __restrict__ bias,
            const float* __restrict__ row_scale, int rs_stride,
            int accumulate)
{
    __shared__ __align__(32) bf16 As[2][128 * 40];   // A tiles, row stride 40
    __shared__ __align__(32) bf16 Bt[2][128 * 48];   // B tiles transposed [n][k], stride 48

    const int tid  = threadIdx.x;
    const int lane = tid & 31;
    const int wv   = tid >> 5;
    const int half = lane >> 4;
    const int l15  = lane & 15;
    const int n0   = blockIdx.x * 128;
    const int m0   = blockIdx.y * 128;

    f32x8 acc[8];
    for (int t = 0; t < 8; ++t) acc[t] = zero8();

    // staging roles
    const int a_row = tid >> 1;          // 0..127
    const int a_kp  = (tid & 1) * 16;    // 0 or 16
    const int b_k   = tid >> 3;          // 0..31
    const int b_ng  = (tid & 7) * 16;    // 0,16,..,112

    auto stageA = [&](int k0, int buf) {
        const bf16* src = A + (size_t)(m0 + a_row) * Kd + k0 + a_kp;
        bf16* dst = &As[buf][a_row * 40 + a_kp];
        async_copy_b128(dst, src);
        async_copy_b128(dst + 8, src + 8);
    };
    auto stageB = [&](int k0, int buf) {
        const float* src = Bm + (size_t)(k0 + b_k) * N + n0 + b_ng;   // coalesced
        bf16* dstcol = &Bt[buf][0];
        for (int j = 0; j < 16; ++j) {
            const float v = (n0 + b_ng + j < N) ? src[j] : 0.f;
            dstcol[(b_ng + j) * 48 + b_k] = (bf16)v;
        }
    };

    const int nk = Kd / 32;
    stageA(0, 0);
    stageB(0, 0);

    for (int ks = 0; ks < nk; ++ks) {
        const int cur = ks & 1;
        async_wait();
        __syncthreads();                       // tile(ks) visible to all waves
        if (ks + 1 < nk) {                     // prefetch next tile into other buffer
            stageA((ks + 1) * 32, cur ^ 1);
            stageB((ks + 1) * 32, cur ^ 1);
        }
        // A fragment (16x32 bf16, ISA lane striping)
        union { bf16x16 v; bf16x8 h[2]; } ua;
        {
            const bf16* p = &As[cur][(wv * 16 + l15) * 40 + half * 8];
            ua.h[0] = *(const bf16x8*)p;         // K = half*8 .. +7
            ua.h[1] = *(const bf16x8*)(p + 16);  // K = 16+half*8 .. +7
        }
        // preload all 8 B fragments, then issue 8 WMMAs
        bf16x16 bfr[8];
        for (int t = 0; t < 8; ++t)
            bfr[t] = [&]{ union { bf16x16 v; bf16x8 h[2]; } ub; const bf16* q = &Bt[cur][(t * 16 + l15) * 48 + half * 8]; ub.h[0] = *(const bf16x8*)q; ub.h[1] = *(const bf16x8*)(q + 16); return ub.v; }();
        for (int t = 0; t < 8; ++t)
            acc[t] = WMMA_BF16(ua.v, bfr[t], acc[t]);
    }

    // epilogue: VGPR r -> rows r / r+8, lanes -> cols
    for (int t = 0; t < 8; ++t) {
        for (int r = 0; r < 8; ++r) {
            const int row = m0 + wv * 16 + r + half * 8;
            const int col = n0 + t * 16 + l15;
            if (col >= N || row >= M) continue;
            float v = acc[t][r];
            if (row_scale) v *= row_scale[(size_t)row * rs_stride];
            if (bias)      v += bias[col];
            if (Cb) {
                Cb[(size_t)row * N + col] = (bf16)v;
            } else {
                float* cp = Cf + (size_t)row * N + col;
                if (accumulate) v += *cp;
                *cp = v;
            }
        }
    }
}

// =====================================================================
// Flash attention, one wave (32 threads) per (b, head, 16-query tile).
// scores = Q Kc^T / 16, softmax online, O = P Vc. All matmuls via WMMA bf16.
// Q/K tiles staged via async global->LDS copies.
// =====================================================================
__global__ __launch_bounds__(32)
void k_attn(const bf16* __restrict__ Qb, const bf16* __restrict__ Kb,
            const bf16* __restrict__ Vb, bf16* __restrict__ Ob)
{
    __shared__ __align__(32) bf16 Qs[16 * 256];
    __shared__ __align__(32) bf16 Ks[32 * 256];
    __shared__ __align__(32) bf16 Vts[256 * 32];  // V^T: [dim][key]
    __shared__ __align__(32) bf16 Ps[16 * 32];

    const int lane = threadIdx.x;
    const int half = lane >> 4, l15 = lane & 15;
    const int qt = blockIdx.x;   // 0..63
    const int hd = blockIdx.y;   // 0..3
    const int b  = blockIdx.z;   // 0..3
    const int tok0 = b * Ss + qt * 16;

    // async-stage 16x256 Q tile for this head
    for (int i = lane; i < 512; i += 32) {
        const int row = i >> 5, cp = (i & 31) * 8;
        async_copy_b128(Qs + row * 256 + cp,
                        Qb + (size_t)(tok0 + row) * Dd + hd * 256 + cp);
    }

    f32x8 accO[16];
    for (int t = 0; t < 16; ++t) accO[t] = zero8();
    float mrow[8], lrow[8];
    for (int r = 0; r < 8; ++r) { mrow[r] = -1e30f; lrow[r] = 0.f; }
    const float scale = 0.0625f;  // 1/sqrt(256)

    for (int kc = 0; kc < Ss; kc += 32) {
        // async-stage K chunk (32 x 256) row-major; it IS B^T layout for Q@K^T
        for (int i = lane; i < 1024; i += 32) {
            const int row = i >> 5, cp = (i & 31) * 8;
            async_copy_b128(Ks + row * 256 + cp,
                            Kb + (size_t)(b * Ss + kc + row) * DK + cp);
        }
        // stage V chunk transposed (needs VGPR round trip)
        for (int i = lane; i < 1024; i += 32) {
            const int row = i >> 5, cp = (i & 31) * 8;
            bf16x8 v = *(const bf16x8*)(Vb + (size_t)(b * Ss + kc + row) * DK + cp);
            for (int j = 0; j < 8; ++j) Vts[(cp + j) * 32 + row] = v[j];
        }
        async_wait();
        __syncthreads();

        // scores 16x32 via 8+8 WMMAs over DK=256
        f32x8 s0 = zero8(), s1 = zero8();
        for (int kk = 0; kk < 256; kk += 32) {
            union { bf16x16 v; bf16x8 h[2]; } ua;
            const bf16* p = Qs + l15 * 256 + kk + half * 8;
            ua.h[0] = *(const bf16x8*)p;
            ua.h[1] = *(const bf16x8*)(p + 16);
            union { bf16x16 v; bf16x8 h[2]; } ub0, ub1;
            const bf16* q0 = Ks + l15 * 256 + kk + half * 8;
            const bf16* q1 = Ks + (16 + l15) * 256 + kk + half * 8;
            ub0.h[0] = *(const bf16x8*)q0; ub0.h[1] = *(const bf16x8*)(q0 + 16);
            ub1.h[0] = *(const bf16x8*)q1; ub1.h[1] = *(const bf16x8*)(q1 + 16);
            bf16x16 b0 = ub0.v;
            bf16x16 b1 = ub1.v;
            s0 = WMMA_BF16(ua.v, b0, s0);
            s1 = WMMA_BF16(ua.v, b1, s1);
        }

        // online softmax per row (16 lanes of a half-wave share a row)
        float fac[8];
        for (int r = 0; r < 8; ++r) {
            float a0 = s0[r] * scale, a1 = s1[r] * scale;
            float mx = fmaxf(a0, a1);
            for (int off = 8; off >= 1; off >>= 1) mx = fmaxf(mx, __shfl_xor(mx, off));
            const float mnew = fmaxf(mrow[r], mx);
            const float f = __expf(mrow[r] - mnew);
            const float p0 = __expf(a0 - mnew), p1 = __expf(a1 - mnew);
            float ps = p0 + p1;
            for (int off = 8; off >= 1; off >>= 1) ps += __shfl_xor(ps, off);
            lrow[r] = lrow[r] * f + ps;
            mrow[r] = mnew;
            fac[r] = f;
            const int prow = r + half * 8;
            Ps[prow * 32 + l15]      = (bf16)p0;
            Ps[prow * 32 + 16 + l15] = (bf16)p1;
        }
        __syncthreads();
        for (int t = 0; t < 16; ++t)
            for (int r = 0; r < 8; ++r) accO[t][r] *= fac[r];

        // O += P(16x32) @ V(32x256): 16 WMMAs
        union { bf16x16 v; bf16x8 h[2]; } up;
        {
            const bf16* p = Ps + l15 * 32 + half * 8;
            up.h[0] = *(const bf16x8*)p;
            up.h[1] = *(const bf16x8*)(p + 16);
        }
        for (int t = 0; t < 16; ++t) {
            union { bf16x16 v; bf16x8 h[2]; } ubv;
            const bf16* qv = Vts + (t * 16 + l15) * 32 + half * 8;
            ubv.h[0] = *(const bf16x8*)qv; ubv.h[1] = *(const bf16x8*)(qv + 16);
            bf16x16 bv = ubv.v;
            accO[t] = WMMA_BF16(up.v, bv, accO[t]);
        }
        __syncthreads();
    }

    // finalize: divide by l, write [B,S,H*DK] layout (bf16 for Wo GEMM)
    for (int t = 0; t < 16; ++t) {
        for (int r = 0; r < 8; ++r) {
            const float v = accO[t][r] / lrow[r];
            const int row = tok0 + r + half * 8;
            const int col = hd * 256 + t * 16 + l15;
            Ob[(size_t)row * Dd + col] = (bf16)v;
        }
    }
}

// =====================================================================
// small kernels
// =====================================================================
__global__ __launch_bounds__(256)
void k_embed(const int* __restrict__ tokens, const float* __restrict__ emb,
             float* __restrict__ X)
{
    const size_t i = (size_t)blockIdx.x * 256 + threadIdx.x;
    const int t = (int)(i / Dd), d = (int)(i % Dd);
    X[i] = emb[(size_t)tokens[t] * Dd + d];
}

__global__ __launch_bounds__(256)
void k_rmsnorm(const float* __restrict__ X, const float* __restrict__ w,
               bf16* __restrict__ out)
{
    const int t = blockIdx.x;
    __shared__ float red[8];
    __shared__ float invs;
    float ss = 0.f;
    for (int d = threadIdx.x; d < Dd; d += 256) {
        const float v = X[(size_t)t * Dd + d];
        ss += v * v;
    }
    for (int off = 16; off >= 1; off >>= 1) ss += __shfl_xor(ss, off);
    if ((threadIdx.x & 31) == 0) red[threadIdx.x >> 5] = ss;
    __syncthreads();
    if (threadIdx.x == 0) {
        float tot = 0.f;
        for (int i = 0; i < 8; ++i) tot += red[i];
        invs = rsqrtf(tot / Dd + EPSc);
    }
    __syncthreads();
    const float inv = invs;
    for (int d = threadIdx.x; d < Dd; d += 256)
        out[(size_t)t * Dd + d] = (bf16)(X[(size_t)t * Dd + d] * inv * w[d]);
}

__global__ __launch_bounds__(256)
void k_router(const float* __restrict__ logits, float* __restrict__ combine)
{
    const int t = blockIdx.x * 8 + (threadIdx.x >> 5);
    const int lane = threadIdx.x & 31;
    float lg = logits[(size_t)t * Ee + lane];
    float mx = lg;
    for (int off = 16; off >= 1; off >>= 1) mx = fmaxf(mx, __shfl_xor(mx, off));
    float p = __expf(lg - mx);
    float s = p;
    for (int off = 16; off >= 1; off >>= 1) s += __shfl_xor(s, off);
    p /= s;
    // iterative top-8 argmax
    bool selected = false;
    float cur = p;
    for (int i = 0; i < Kk; ++i) {
        float mv = cur;
        for (int off = 16; off >= 1; off >>= 1) mv = fmaxf(mv, __shfl_xor(mv, off));
        const unsigned long long bal = __ballot(cur == mv);
        const int first = __ffsll((unsigned long long)bal) - 1;
        if (lane == first) { selected = true; cur = -1.f; }
    }
    // renormalize top-k probs via softmax over selected probabilities
    float tm = selected ? p : -1e30f;
    for (int off = 16; off >= 1; off >>= 1) tm = fmaxf(tm, __shfl_xor(tm, off));
    float se = selected ? __expf(p - tm) : 0.f;
    float ssum = se;
    for (int off = 16; off >= 1; off >>= 1) ssum += __shfl_xor(ssum, off);
    combine[(size_t)t * Ee + lane] = selected ? (se / ssum) : 0.f;
}

__global__ __launch_bounds__(256)
void k_silumul(const float* __restrict__ a, const float* __restrict__ b,
               bf16* __restrict__ out)
{
    const size_t i = (size_t)blockIdx.x * 256 + threadIdx.x;
    const float x = a[i];
    out[i] = (bf16)((x / (1.f + __expf(-x))) * b[i]);
}

__global__ __launch_bounds__(256)
void k_axpy(float* __restrict__ X, const float* __restrict__ Y)
{
    const size_t i = (size_t)blockIdx.x * 256 + threadIdx.x;
    X[i] += Y[i];
}

__global__ __launch_bounds__(256)
void k_invrms(const float* __restrict__ X, float* __restrict__ invr)
{
    const int t = blockIdx.x;
    __shared__ float red[8];
    float ss = 0.f;
    for (int d = threadIdx.x; d < Dd; d += 256) {
        const float v = X[(size_t)t * Dd + d];
        ss += v * v;
    }
    for (int off = 16; off >= 1; off >>= 1) ss += __shfl_xor(ss, off);
    if ((threadIdx.x & 31) == 0) red[threadIdx.x >> 5] = ss;
    __syncthreads();
    if (threadIdx.x == 0) {
        float tot = 0.f;
        for (int i = 0; i < 8; ++i) tot += red[i];
        invr[t] = rsqrtf(tot / Dd + EPSc);
    }
}

__global__ __launch_bounds__(256)
void k_pool(const float* __restrict__ X, const float* __restrict__ invr,
            const float* __restrict__ fw, float* __restrict__ pooled)
{
    const int b = blockIdx.y;
    const int d = blockIdx.x * 256 + threadIdx.x;
    float s = 0.f;
    for (int srow = 0; srow < Ss; ++srow)
        s += X[(size_t)(b * Ss + srow) * Dd + d] * invr[b * Ss + srow];
    pooled[(size_t)b * Dd + d] = s * fw[d] * (1.f / Ss);
}

__global__ __launch_bounds__(256)
void k_cls(const float* __restrict__ pooled, const float* __restrict__ cw,
           const float* __restrict__ cb, float* __restrict__ out)
{
    const int b = blockIdx.x;
    __shared__ float lg[NCc];
    __shared__ float red[8];
    for (int c = 0; c < NCc; ++c) {
        float s = 0.f;
        for (int d = threadIdx.x; d < Dd; d += 256)
            s += pooled[(size_t)b * Dd + d] * cw[(size_t)d * NCc + c];
        for (int off = 16; off >= 1; off >>= 1) s += __shfl_xor(s, off);
        if ((threadIdx.x & 31) == 0) red[threadIdx.x >> 5] = s;
        __syncthreads();
        if (threadIdx.x == 0) {
            float tot = 0.f;
            for (int i = 0; i < 8; ++i) tot += red[i];
            lg[c] = tot + cb[c];
        }
        __syncthreads();
    }
    if (threadIdx.x == 0) {
        float mx = lg[0];
        for (int c = 1; c < NCc; ++c) mx = fmaxf(mx, lg[c]);
        float s = 0.f, e[NCc];
        for (int c = 0; c < NCc; ++c) { e[c] = __expf(lg[c] - mx); s += e[c]; }
        for (int c = 0; c < NCc; ++c) out[(size_t)b * NCc + c] = e[c] / s;
    }
}

// =====================================================================
// host orchestration
// =====================================================================
extern "C" void kernel_launch(void* const* d_in, const int* in_sizes, int n_in,
                              void* d_out, int out_size, void* d_ws, size_t ws_size,
                              hipStream_t stream)
{
    (void)in_sizes; (void)n_in; (void)out_size; (void)ws_size;

    const int*   tokens     = (const int*)  d_in[0];
    const float* emb        = (const float*)d_in[1];
    const float* norm1_w    = (const float*)d_in[2];
    const float* norm2_w    = (const float*)d_in[3];
    const float* Wq         = (const float*)d_in[4];
    const float* Wk         = (const float*)d_in[5];
    const float* Wv         = (const float*)d_in[6];
    const float* Wo         = (const float*)d_in[7];
    const float* router_w   = (const float*)d_in[8];
    const float* exp_bias   = (const float*)d_in[9];
    const float* sh_w1      = (const float*)d_in[10];
    const float* sh_w3      = (const float*)d_in[11];
    const float* sh_w2      = (const float*)d_in[12];
    const float* ex_w1      = (const float*)d_in[13];
    const float* ex_w3      = (const float*)d_in[14];
    const float* ex_w2      = (const float*)d_in[15];
    const float* final_w    = (const float*)d_in[16];
    const float* cls_w      = (const float*)d_in[17];
    const float* cls_b      = (const float*)d_in[18];
    float* out = (float*)d_out;

    // ---- workspace carve-out (256B aligned) ----
    size_t off = 0;
    auto alloc = [&](size_t bytes) -> void* {
        void* p = (char*)d_ws + off;
        off += (bytes + 255) & ~(size_t)255;
        return p;
    };
    float* Xf      = (float*)alloc((size_t)Tt * Dd * 4);
    bf16*  h1      = (bf16*) alloc((size_t)Tt * Dd * 2);
    bf16*  h2      = (bf16*) alloc((size_t)Tt * Dd * 2);
    bf16*  Qb      = (bf16*) alloc((size_t)Tt * Dd * 2);
    bf16*  Kb      = (bf16*) alloc((size_t)Tt * DK * 2);
    bf16*  Vb      = (bf16*) alloc((size_t)Tt * DK * 2);
    bf16*  AO      = (bf16*) alloc((size_t)Tt * Dd * 2);
    float* logits  = (float*)alloc((size_t)Tt * Ee * 4);
    float* combine = (float*)alloc((size_t)Tt * Ee * 4);
    float* tmp1    = (float*)alloc((size_t)Tt * Ff * 4);
    float* tmp2    = (float*)alloc((size_t)Tt * Ff * 4);
    bf16*  act     = (bf16*) alloc((size_t)Tt * Ff * 2);
    float* spec    = (float*)alloc((size_t)Tt * Dd * 4);
    float* invr    = (float*)alloc((size_t)Tt * 4);
    float* pooled  = (float*)alloc((size_t)Bb * Dd * 4);

    const dim3 blk256(256), blk32(32);

    // 1) embedding gather
    k_embed<<<Tt * Dd / 256, blk256, 0, stream>>>(tokens, emb, Xf);
    // 2) rmsnorm -> h1 (bf16)
    k_rmsnorm<<<Tt, blk256, 0, stream>>>(Xf, norm1_w, h1);
    // 3) Q/K/V projections (bf16 outputs). Grid: (ceil(N/128), M/128)
    k_gemm<<<dim3(8, 32), blk256, 0, stream>>>(h1, Wq, nullptr, Qb, Tt, Dd, Dd, nullptr, nullptr, 0, 0);
    k_gemm<<<dim3(2, 32), blk256, 0, stream>>>(h1, Wk, nullptr, Kb, Tt, DK, Dd, nullptr, nullptr, 0, 0);
    k_gemm<<<dim3(2, 32), blk256, 0, stream>>>(h1, Wv, nullptr, Vb, Tt, DK, Dd, nullptr, nullptr, 0, 0);
    // 4) flash attention (B,H,Stiles)
    k_attn<<<dim3(Ss / 16, Hh, Bb), blk32, 0, stream>>>(Qb, Kb, Vb, AO);
    // 5) output projection, residual accumulate into Xf
    k_gemm<<<dim3(8, 32), blk256, 0, stream>>>(AO, Wo, Xf, nullptr, Tt, Dd, Dd, nullptr, nullptr, 0, 1);
    // 6) second rmsnorm -> h2
    k_rmsnorm<<<Tt, blk256, 0, stream>>>(Xf, norm2_w, h2);
    // 7) router logits (N=32, col-guarded tile) + top-k combine weights
    k_gemm<<<dim3(1, 32), blk256, 0, stream>>>(h2, router_w, logits, nullptr, Tt, Ee, Dd, exp_bias, nullptr, 0, 0);
    k_router<<<Tt / 8, blk256, 0, stream>>>(logits, combine);
    // 8) shared expert -> initializes spec
    k_gemm<<<dim3(4, 32), blk256, 0, stream>>>(h2, sh_w1, tmp1, nullptr, Tt, Ff, Dd, nullptr, nullptr, 0, 0);
    k_gemm<<<dim3(4, 32), blk256, 0, stream>>>(h2, sh_w3, tmp2, nullptr, Tt, Ff, Dd, nullptr, nullptr, 0, 0);
    k_silumul<<<Tt * Ff / 256, blk256, 0, stream>>>(tmp1, tmp2, act);
    k_gemm<<<dim3(8, 32), blk256, 0, stream>>>(act, sh_w2, spec, nullptr, Tt, Dd, Ff, nullptr, nullptr, 0, 0);
    // 9) routed experts: spec += combine[:,e] * (silu(h2 w1_e) * (h2 w3_e)) w2_e
    for (int e = 0; e < Ee; ++e) {
        const float* w1e = ex_w1 + (size_t)e * Dd * Ff;
        const float* w3e = ex_w3 + (size_t)e * Dd * Ff;
        const float* w2e = ex_w2 + (size_t)e * Ff * Dd;
        k_gemm<<<dim3(4, 32), blk256, 0, stream>>>(h2, w1e, tmp1, nullptr, Tt, Ff, Dd, nullptr, nullptr, 0, 0);
        k_gemm<<<dim3(4, 32), blk256, 0, stream>>>(h2, w3e, tmp2, nullptr, Tt, Ff, Dd, nullptr, nullptr, 0, 0);
        k_silumul<<<Tt * Ff / 256, blk256, 0, stream>>>(tmp1, tmp2, act);
        k_gemm<<<dim3(8, 32), blk256, 0, stream>>>(act, w2e, spec, nullptr, Tt, Dd, Ff,
                                                   nullptr, combine + e, Ee, 1);
    }
    // 10) residual add
    k_axpy<<<Tt * Dd / 256, blk256, 0, stream>>>(Xf, spec);
    // 11) final norm + mean pool + classifier softmax
    k_invrms<<<Tt, blk256, 0, stream>>>(Xf, invr);
    k_pool<<<dim3(Dd / 256, Bb), blk256, 0, stream>>>(Xf, invr, final_w, pooled);
    k_cls<<<Bb, blk256, 0, stream>>>(pooled, cls_w, cls_b, out);
}
